// CIFARViT_59030030516906
// MI455X (gfx1250) — matrix operand order repaired and checked by reading, not compiled, not run on hardware
//
#include <hip/hip_runtime.h>
#include <hip/hip_bf16.h>
#include <math.h>

// ---------------------------------------------------------------------------
// Types for CDNA5 WMMA (wave32): v_wmma_f32_16x16x32_bf16
// ---------------------------------------------------------------------------
typedef __bf16 bf16_t;
typedef __attribute__((ext_vector_type(16))) __bf16 v16bf;
typedef __attribute__((ext_vector_type(8)))  __bf16 v8bf;
typedef __attribute__((ext_vector_type(8)))  float  v8f;

#define ACT_NONE 0
#define ACT_GELU 1

__device__ __forceinline__ bf16_t f2bf(float f) {
  unsigned int u = __float_as_uint(f);
  u += 0x7FFFu + ((u >> 16) & 1u);          // round-to-nearest-even
  unsigned short hs = (unsigned short)(u >> 16);
  bf16_t r;
  __builtin_memcpy(&r, &hs, sizeof(r));
  return r;
}

__device__ __forceinline__ float gelu_exact(float x) {
  return 0.5f * x * (1.0f + erff(x * 0.70710678118654752f));
}

// ---------------------------------------------------------------------------
// Batched WMMA GEMM (guard-free):  C[z] = act(alpha*A[z]@B[z]^T + bias) (+=C)
//   A row-major [M,K] (lda), B stored weight-style [N,K] (ldb).
//   Wave tile: 16 (M) x 16*NSUB (N): NSUB accumulators share one A fragment.
//   Block: 128 threads = 4 waves stacked in M -> 64 x 16*NSUB block tile.
//   REQUIREMENT (enforced by launch sites): M % 64 == 0, N % (16*NSUB) == 0,
//   K % 32 == 0. No bounds checks -> no EXEC juggling in the hot loop:
//   2*NSUB+2 b128 vector loads, clauses, NSUB back-to-back v_wmma per k-step.
//   WBF16: epilogue additionally stores a bf16 copy of C (fused downconvert,
//   saves a full f32 read + bf16 write pass per activation tensor).
// Batch offset: z1 = z / zdiv, z2 = z % zdiv; off = z1*s?1 + z2*s?2.
// A/B fragments use contiguous k = 16*(lane/16) + s (same in both); f32 C/D per CDNA5 ISA 7.12.2.
// ---------------------------------------------------------------------------
template<int ACT, int ACC, int BIAS, int NSUB, int WBF16>
__global__ void __launch_bounds__(128)
k_gemm(const bf16_t* __restrict__ A, const bf16_t* __restrict__ Bm,
       const float* __restrict__ bias, float* __restrict__ C,
       bf16_t* __restrict__ C16,
       int K, int lda, int ldb, int ldc,
       long long sA1, long long sA2, long long sB1, long long sB2,
       long long sC1, long long sC2, int zdiv, float alpha)
{
  const int z  = blockIdx.z;
  const int z1 = z / zdiv;
  const int z2 = z % zdiv;
  const bf16_t* Ab = A  + z1 * sA1 + z2 * sA2;
  const bf16_t* Bb = Bm + z1 * sB1 + z2 * sB2;
  float*        Cb = C  + z1 * sC1 + z2 * sC2;
  bf16_t*       Hb = WBF16 ? (C16 + z1 * sC1 + z2 * sC2) : nullptr;

  const int lane = threadIdx.x & 31;
  const int wid  = threadIdx.x >> 5;
  const int tM   = blockIdx.y * 64 + wid * 16;
  const int tN0  = blockIdx.x * (16 * NSUB);

  const int mrow = lane & 15;
  const int hi   = lane >> 4;               // lane half selects K sub-range

  const bf16_t* pa = Ab + (long long)(tM + mrow) * lda + hi * 16;
  const bf16_t* pb[NSUB];
#pragma unroll
  for (int j = 0; j < NSUB; ++j)
    pb[j] = Bb + (long long)(tN0 + 16 * j + mrow) * ldb + hi * 16;

  v8f acc[NSUB];
#pragma unroll
  for (int j = 0; j < NSUB; ++j)
#pragma unroll
    for (int r = 0; r < 8; ++r) acc[j][r] = 0.0f;

  for (int k0 = 0; k0 < K; k0 += 32) {
    v16bf a;
    {
      v8bf lo = *(const v8bf*)(pa + k0);
      v8bf hh = *(const v8bf*)(pa + k0 + 8);
#pragma unroll
      for (int i = 0; i < 8; ++i) { a[i] = lo[i]; a[i + 8] = hh[i]; }
    }
    v16bf b[NSUB];
#pragma unroll
    for (int j = 0; j < NSUB; ++j)
      b[j] = *(const v16bf*)(pb[j] + k0);
#pragma unroll
    for (int j = 0; j < NSUB; ++j)
      acc[j] = __builtin_amdgcn_wmma_f32_16x16x32_bf16(
          false, a, false, b[j], (short)0, acc[j], false, false);
  }

  // epilogue: VGPR r -> (row = tM + r + 8*hi, col = subtile base + lane&15)
#pragma unroll
  for (int j = 0; j < NSUB; ++j) {
    const int col = tN0 + 16 * j + mrow;
    float bv = 0.0f;
    if (BIAS) bv = bias[col];
#pragma unroll
    for (int r = 0; r < 8; ++r) {
      const int row = tM + r + 8 * hi;
      float v = acc[j][r] * alpha;
      if (BIAS) v += bv;
      if (ACT == ACT_GELU) v = gelu_exact(v);
      const long long idx = (long long)row * ldc + col;
      if (ACC) v += Cb[idx];
      Cb[idx] = v;
      if (WBF16) Hb[idx] = f2bf(v);
    }
  }
}

template<int ACT, int ACC, int BIAS, int NSUB, int WBF16>
static inline void gemm(hipStream_t s, const bf16_t* A, const bf16_t* B,
                        const float* bias, float* C, bf16_t* C16,
                        int M, int N, int K, int lda, int ldb, int ldc,
                        long long sA1, long long sA2, long long sB1, long long sB2,
                        long long sC1, long long sC2, int Z, int zdiv, float alpha)
{
  dim3 g((unsigned)(N / (16 * NSUB)), (unsigned)(M / 64), (unsigned)Z);
  k_gemm<ACT, ACC, BIAS, NSUB, WBF16><<<g, 128, 0, s>>>(
      A, B, bias, C, C16, K, lda, ldb, ldc,
      sA1, sA2, sB1, sB2, sC1, sC2, zdiv, alpha);
}

// ---------------------------------------------------------------------------
// LDS-tiled batched transpose with f32 -> bf16 downconvert.
// Input (per z): in[r, s] = in[z1*si1 + z2*si2 + r*ldi + s], r<NR, s<NS.
// Output (per z): out[z*so + s*NR + r]  (i.e. [NS, NR] row-major, "weight" form)
// ---------------------------------------------------------------------------
__global__ void __launch_bounds__(256)
k_transpose(const float* __restrict__ in, bf16_t* __restrict__ out,
            int NR, int NS, int ldi, long long si1, long long si2,
            long long so, int zdiv)
{
  __shared__ float tile[32][33];
  const int z = blockIdx.z, z1 = z / zdiv, z2 = z % zdiv;
  const float* ib = in + z1 * si1 + z2 * si2;
  bf16_t* ob = out + (long long)z * so;
  const int r0 = blockIdx.y * 32, s0 = blockIdx.x * 32;
  const int tx = threadIdx.x & 31, ty = threadIdx.x >> 5;   // 32 x 8
  for (int rr = ty; rr < 32; rr += 8) {
    const int r = r0 + rr, s = s0 + tx;
    tile[tx][rr] = (r < NR && s < NS) ? ib[(long long)r * ldi + s] : 0.0f;
  }
  __syncthreads();
  for (int ss = ty; ss < 32; ss += 8) {
    const int s = s0 + ss, r = r0 + tx;
    if (s < NS && r < NR) ob[(long long)s * NR + r] = f2bf(tile[ss][tx]);
  }
}

// ---------------------------------------------------------------------------
// Elementwise / reduction kernels (fp32 VALU)
// ---------------------------------------------------------------------------
__global__ void __launch_bounds__(256)
k_cvt(const float* __restrict__ s, bf16_t* __restrict__ d, long long n)
{
  long long i = (long long)blockIdx.x * 256 + threadIdx.x;
  if (i < n) d[i] = f2bf(s[i]);
}

__global__ void __launch_bounds__(256)
k_add(float* __restrict__ dst, const float* __restrict__ src, long long n)
{
  long long i = (long long)blockIdx.x * 256 + threadIdx.x;
  if (i < n) dst[i] += src[i];
}

// LayerNorm over last dim C; one row per block; bf16 out (+ optional f32 out).
__global__ void __launch_bounds__(256)
k_layernorm(const float* __restrict__ x, const float* __restrict__ g,
            const float* __restrict__ b, bf16_t* __restrict__ y,
            float* __restrict__ yf, int C, float eps)
{
  const long long row = blockIdx.x;
  const float* xr = x + row * C;
  bf16_t* yr = y + row * C;
  __shared__ float s1[256], s2[256];
  float sum = 0.0f, sq = 0.0f;
  for (int i = threadIdx.x; i < C; i += 256) { float v = xr[i]; sum += v; sq += v * v; }
  s1[threadIdx.x] = sum; s2[threadIdx.x] = sq; __syncthreads();
  for (int off = 128; off > 0; off >>= 1) {
    if ((int)threadIdx.x < off) {
      s1[threadIdx.x] += s1[threadIdx.x + off];
      s2[threadIdx.x] += s2[threadIdx.x + off];
    }
    __syncthreads();
  }
  const float mu = s1[0] / (float)C;
  const float var = s2[0] / (float)C - mu * mu;
  const float rs = rsqrtf(var + eps);
  for (int i = threadIdx.x; i < C; i += 256) {
    const float v = (xr[i] - mu) * rs * g[i] + b[i];
    yr[i] = f2bf(v);
    if (yf) yf[row * C + i] = v;
  }
}

// Row softmax over last dim N; writes f32 (may alias input) and bf16.
__global__ void __launch_bounds__(256)
k_softmax_row(const float* __restrict__ x, float* __restrict__ yf,
              bf16_t* __restrict__ yb, int N)
{
  const long long row = blockIdx.x;
  const float* xr = x + row * N;
  __shared__ float s[256];
  float m = -1e30f;
  for (int i = threadIdx.x; i < N; i += 256) m = fmaxf(m, xr[i]);
  s[threadIdx.x] = m; __syncthreads();
  for (int off = 128; off > 0; off >>= 1) {
    if ((int)threadIdx.x < off) s[threadIdx.x] = fmaxf(s[threadIdx.x], s[threadIdx.x + off]);
    __syncthreads();
  }
  m = s[0]; __syncthreads();
  float sum = 0.0f;
  for (int i = threadIdx.x; i < N; i += 256) sum += __expf(xr[i] - m);
  s[threadIdx.x] = sum; __syncthreads();
  for (int off = 128; off > 0; off >>= 1) {
    if ((int)threadIdx.x < off) s[threadIdx.x] += s[threadIdx.x + off];
    __syncthreads();
  }
  const float inv = 1.0f / s[0];
  for (int i = threadIdx.x; i < N; i += 256) {
    const float e = __expf(xr[i] - m) * inv;
    yf[row * N + i] = e;
    yb[row * N + i] = f2bf(e);
  }
}

// Softmax over the middle dim of [Z, Lr, Mm] (axis=-2), output TRANSPOSED:
// y[z, m, l] = softmax_l(x[z, l, m])  -> bf16, directly usable as GEMM A.
__global__ void __launch_bounds__(256)
k_softmax_mid_t(const float* __restrict__ x, bf16_t* __restrict__ y, int Lr, int Mm)
{
  const int m = blockIdx.x * 256 + threadIdx.x;
  if (m >= Mm) return;
  const float* base = x + (long long)blockIdx.y * Lr * Mm + m;
  bf16_t* yb = y + ((long long)blockIdx.y * Mm + m) * Lr;
  float mx = -1e30f;
  for (int l = 0; l < Lr; ++l) mx = fmaxf(mx, base[(long long)l * Mm]);
  float s = 0.0f;
  for (int l = 0; l < Lr; ++l) s += __expf(base[(long long)l * Mm] - mx);
  const float inv = 1.0f / s;
  for (int l = 0; l < Lr; ++l)
    yb[l] = f2bf(__expf(base[(long long)l * Mm] - mx) * inv);
}

// Patch embed: 2x2/s2 conv on [8,3,32,32] + bias + pos_emb -> h [8,256,768]
__global__ void __launch_bounds__(256)
k_patch(const float* __restrict__ x, const float* __restrict__ w,
        const float* __restrict__ pb, const float* __restrict__ pos,
        float* __restrict__ h)
{
  const int idx = blockIdx.x * 256 + threadIdx.x;
  if (idx >= 8 * 256 * 768) return;
  const int c = idx % 768;
  const int l = (idx / 768) % 256;
  const int b = idx / (768 * 256);
  const int py = l >> 4, px = l & 15;
  float acc = pb[c];
#pragma unroll
  for (int ci = 0; ci < 3; ++ci)
#pragma unroll
    for (int ky = 0; ky < 2; ++ky)
#pragma unroll
      for (int kx = 0; kx < 2; ++kx)
        acc += x[((b * 3 + ci) * 32 + (2 * py + ky)) * 32 + (2 * px + kx)]
             * w[((c * 3 + ci) * 2 + ky) * 2 + kx];
  h[idx] = acc + pos[l * 768 + c];
}

// Mean over L of h [8,256,768] -> [8,768]
__global__ void __launch_bounds__(256)
k_pool(const float* __restrict__ h, float* __restrict__ out)
{
  const int idx = blockIdx.x * 256 + threadIdx.x;
  if (idx >= 8 * 768) return;
  const int c = idx % 768, b = idx / 768;
  float s = 0.0f;
  for (int l = 0; l < 256; ++l) s += h[((long long)b * 256 + l) * 768 + c];
  out[idx] = s * (1.0f / 256.0f);
}

// Tiny ragged head GEMM: out[8,10] = ln_pooled[8,768] @ head_w[10,768]^T + b
__global__ void __launch_bounds__(128)
k_head(const float* __restrict__ xin, const float* __restrict__ w,
       const float* __restrict__ b, float* __restrict__ out)
{
  const int t = threadIdx.x;
  if (t >= 80) return;
  const int bb = t / 10, n = t % 10;
  float s = b[n];
  for (int k = 0; k < 768; ++k) s += xin[bb * 768 + k] * w[n * 768 + k];
  out[t] = s;
}

// ---------------------------------------------------------------------------
// Orchestration
// ---------------------------------------------------------------------------
extern "C" void kernel_launch(void* const* d_in, const int* in_sizes, int n_in,
                              void* d_out, int out_size, void* d_ws, size_t ws_size,
                              hipStream_t stream)
{
  (void)in_sizes; (void)n_in; (void)out_size; (void)ws_size;
  const float* x        = (const float*)d_in[0];
  const float* patch_w  = (const float*)d_in[1];
  const float* patch_b  = (const float*)d_in[2];
  const float* pos_emb  = (const float*)d_in[3];
  const float* ln1_g    = (const float*)d_in[4];
  const float* ln1_b    = (const float*)d_in[5];
  const float* qkv_w    = (const float*)d_in[6];
  const float* proj_w   = (const float*)d_in[7];
  const float* proj_b   = (const float*)d_in[8];
  const float* ln2_g    = (const float*)d_in[9];
  const float* ln2_b    = (const float*)d_in[10];
  const float* mlp_w1   = (const float*)d_in[11];
  const float* mlp_b1   = (const float*)d_in[12];
  const float* mlp_w2   = (const float*)d_in[13];
  const float* mlp_b2   = (const float*)d_in[14];
  const float* m_ln1_g  = (const float*)d_in[15];
  const float* m_ln1_b  = (const float*)d_in[16];
  const float* m_qkv_w  = (const float*)d_in[17];
  const float* m_proj_w = (const float*)d_in[18];
  const float* m_proj_b = (const float*)d_in[19];
  const float* m_ln2_g  = (const float*)d_in[20];
  const float* m_ln2_b  = (const float*)d_in[21];
  const float* m_mlp_w1 = (const float*)d_in[22];
  const float* m_mlp_b1 = (const float*)d_in[23];
  const float* m_mlp_w2 = (const float*)d_in[24];
  const float* m_mlp_b2 = (const float*)d_in[25];
  const float* norm_g   = (const float*)d_in[26];
  const float* norm_b   = (const float*)d_in[27];
  const float* head_w   = (const float*)d_in[28];
  const float* head_b   = (const float*)d_in[29];

  const int Bx = 8, Lx = 256, Cc = 768, Hh = 8, hd = 96, Dd = 8, Ff = 3072;
  const int C3 = 3 * Cc;          // 2304
  const int BH = Bx * Hh;         // 64
  const int ML = 256, MC3 = 768, MF = 1024;
  const int rowsO = Bx * Lx;      // 2048
  const int rowsM = BH * ML;      // 16384

  char* cur = (char*)d_ws;
  auto walloc = [&](size_t bytes) -> void* {
    void* r = (void*)cur;
    cur += (bytes + 255) & ~((size_t)255);
    return r;
  };

  float*  h_f       = (float*) walloc((size_t)rowsO * Cc * 4);
  bf16_t* bf1       = (bf16_t*)walloc((size_t)rowsO * Cc * 2);
  float*  qkv_f     = (float*) walloc((size_t)rowsO * C3 * 4);
  bf16_t* qkv_bf    = (bf16_t*)walloc((size_t)rowsO * C3 * 2);
  float*  sc_f      = (float*) walloc((size_t)BH * Lx * Lx * 4);
  bf16_t* a_bf      = (bf16_t*)walloc((size_t)BH * Lx * Lx * 2);
  bf16_t* vT_bf     = (bf16_t*)walloc((size_t)BH * hd * Lx * 2);
  float*  ao_f      = (float*) walloc((size_t)rowsO * Cc * 4);
  bf16_t* ao_bf     = (bf16_t*)walloc((size_t)rowsO * Cc * 2);
  float*  op_f      = (float*) walloc((size_t)rowsO * Cc * 4);
  bf16_t* opT_bf    = (bf16_t*)walloc((size_t)BH * hd * Lx * 2);
  bf16_t* mbf1      = (bf16_t*)walloc((size_t)rowsM * ML * 2);
  float*  mqkv_f    = (float*) walloc((size_t)rowsM * MC3 * 4);
  bf16_t* mqkv_bf   = (bf16_t*)walloc((size_t)rowsM * MC3 * 2);
  bf16_t* mvT_bf    = (bf16_t*)walloc((size_t)rowsM * ML * 2);
  float*  msc_f     = (float*) walloc((size_t)rowsM * ML * 4);
  bf16_t* ms_bf     = (bf16_t*)walloc((size_t)rowsM * ML * 2);
  float*  mo_f      = (float*) walloc((size_t)rowsM * ML * 4);
  bf16_t* mo_bf     = (bf16_t*)walloc((size_t)rowsM * ML * 2);
  float*  mh_f      = (float*) walloc((size_t)rowsM * ML * 4);
  float*  mh1_f     = (float*) walloc((size_t)rowsM * MF * 4);
  bf16_t* mh1_bf    = (bf16_t*)walloc((size_t)rowsM * MF * 2);
  bf16_t* wnT_bf    = (bf16_t*)walloc((size_t)BH * Lx * Lx * 2);
  float*  h1_f      = (float*) walloc((size_t)rowsO * Ff * 4);
  bf16_t* h1_bf     = (bf16_t*)walloc((size_t)rowsO * Ff * 2);
  float*  pool_f    = (float*) walloc((size_t)Bx * Cc * 4);
  bf16_t* pool_bf   = (bf16_t*)walloc((size_t)Bx * Cc * 2);
  float*  pooln_f   = (float*) walloc((size_t)Bx * Cc * 4);
  bf16_t* qkvw_bf   = (bf16_t*)walloc((size_t)C3 * Cc * 2);
  bf16_t* projw_bf  = (bf16_t*)walloc((size_t)Cc * Cc * 2);
  bf16_t* w1_bf     = (bf16_t*)walloc((size_t)Ff * Cc * 2);
  bf16_t* w2_bf     = (bf16_t*)walloc((size_t)Cc * Ff * 2);
  bf16_t* mqkvw_bf  = (bf16_t*)walloc((size_t)MC3 * ML * 2);
  bf16_t* mprojw_bf = (bf16_t*)walloc((size_t)ML * ML * 2);
  bf16_t* mw1_bf    = (bf16_t*)walloc((size_t)MF * ML * 2);
  bf16_t* mw2_bf    = (bf16_t*)walloc((size_t)ML * MF * 2);

  auto cvt = [&](const float* s, bf16_t* d, long long n) {
    k_cvt<<<dim3((unsigned)((n + 255) / 256)), 256, 0, stream>>>(s, d, n);
  };

  // --- mini weights to bf16 (once per launch) ---
  cvt(m_qkv_w,  mqkvw_bf,  (long long)MC3 * ML);
  cvt(m_proj_w, mprojw_bf, (long long)ML * ML);
  cvt(m_mlp_w1, mw1_bf,    (long long)MF * ML);
  cvt(m_mlp_w2, mw2_bf,    (long long)ML * MF);

  // --- patch embed + pos ---
  k_patch<<<dim3((unsigned)((rowsO * Cc + 255) / 256)), 256, 0, stream>>>(
      x, patch_w, patch_b, pos_emb, h_f);

  const float sc_outer = 1.0f / sqrtf((float)hd);
  const float sc_mini  = 1.0f / sqrtf((float)ML);

  for (int i = 0; i < Dd; ++i) {
    cvt(qkv_w  + (size_t)i * C3 * Cc, qkvw_bf,  (long long)C3 * Cc);
    cvt(proj_w + (size_t)i * Cc * Cc, projw_bf, (long long)Cc * Cc);
    cvt(mlp_w1 + (size_t)i * Ff * Cc, w1_bf,    (long long)Ff * Cc);
    cvt(mlp_w2 + (size_t)i * Cc * Ff, w2_bf,    (long long)Cc * Ff);

    // hn = LN(h); qkv = hn @ qkv_w^T  (f32 + fused bf16 out)
    k_layernorm<<<rowsO, 256, 0, stream>>>(h_f, ln1_g + (size_t)i * Cc,
                                           ln1_b + (size_t)i * Cc, bf1, nullptr,
                                           Cc, 1e-6f);
    gemm<ACT_NONE, 0, 0, 4, 1>(stream, bf1, qkvw_bf, nullptr, qkv_f, qkv_bf,
        rowsO, C3, Cc, Cc, Cc, C3, 0, 0, 0, 0, 0, 0, 1, 1, 1.0f);
    // v^T[b,h,d,m] from qkv (for a@v as NTxT GEMM)
    k_transpose<<<dim3(3, 8, BH), 256, 0, stream>>>(
        qkv_f + 2 * Cc, vT_bf, Lx, hd, C3,
        (long long)Lx * C3, hd, (long long)hd * Lx, Hh);

    // scores[b,h] = scale * q @ k^T   (z = b*H + h)
    gemm<ACT_NONE, 0, 0, 4, 0>(stream, qkv_bf, qkv_bf + Cc, nullptr, sc_f, nullptr,
        Lx, Lx, hd, C3, C3, Lx,
        (long long)Lx * C3, hd, (long long)Lx * C3, hd,
        (long long)Hh * Lx * Lx, (long long)Lx * Lx, BH, Hh, sc_outer);
    k_softmax_row<<<BH * Lx, 256, 0, stream>>>(sc_f, sc_f, a_bf, Lx);

    // attn out = a @ v  -> ao [B,L,H,hd]=[B,L,C]   (N = 96 -> NSUB = 2)
    gemm<ACT_NONE, 0, 0, 2, 1>(stream, a_bf, vT_bf, nullptr, ao_f, ao_bf,
        Lx, hd, Lx, Lx, Lx, Cc,
        (long long)Hh * Lx * Lx, (long long)Lx * Lx,
        (long long)Hh * hd * Lx, (long long)hd * Lx,
        (long long)Lx * Cc, hd, BH, Hh, 1.0f);

    // o = ao @ proj_w^T + proj_b ; then o^T per head for the wn-mixing GEMM
    gemm<ACT_NONE, 0, 1, 4, 0>(stream, ao_bf, projw_bf,
        proj_b + (size_t)i * Cc, op_f, nullptr,
        rowsO, Cc, Cc, Cc, Cc, Cc, 0, 0, 0, 0, 0, 0, 1, 1, 1.0f);
    k_transpose<<<dim3(3, 8, BH), 256, 0, stream>>>(
        op_f, opT_bf, Lx, hd, Cc,
        (long long)Lx * Cc, hd, (long long)hd * Lx, Hh);

    // ---------------- mini ViT on attention maps (w = sc_f probs) ----------
    k_layernorm<<<rowsM, 256, 0, stream>>>(sc_f, m_ln1_g, m_ln1_b, mbf1, nullptr,
                                           ML, 1e-6f);
    gemm<ACT_NONE, 0, 0, 4, 1>(stream, mbf1, mqkvw_bf, nullptr, mqkv_f, mqkv_bf,
        rowsM, MC3, ML, ML, ML, MC3, 0, 0, 0, 0, 0, 0, 1, 1, 1.0f);
    k_transpose<<<dim3(8, 8, BH), 256, 0, stream>>>(
        mqkv_f + 2 * ML, mvT_bf, ML, ML, MC3,
        (long long)ML * MC3, 0, (long long)ML * ML, 1);

    gemm<ACT_NONE, 0, 0, 4, 0>(stream, mqkv_bf, mqkv_bf + ML, nullptr, msc_f, nullptr,
        ML, ML, ML, MC3, MC3, ML,
        (long long)ML * MC3, 0, (long long)ML * MC3, 0,
        (long long)ML * ML, 0, BH, 1, sc_mini);
    k_softmax_row<<<rowsM, 256, 0, stream>>>(msc_f, msc_f, ms_bf, ML);

    gemm<ACT_NONE, 0, 0, 4, 1>(stream, ms_bf, mvT_bf, nullptr, mo_f, mo_bf,
        ML, ML, ML, ML, ML, ML,
        (long long)ML * ML, 0, (long long)ML * ML, 0,
        (long long)ML * ML, 0, BH, 1, 1.0f);

    // mh = mo @ pw^T + pb ; mh += w @ pw^T + pb (residual through proj)
    gemm<ACT_NONE, 0, 1, 4, 0>(stream, mo_bf, mprojw_bf, m_proj_b, mh_f, nullptr,
        rowsM, ML, ML, ML, ML, ML, 0, 0, 0, 0, 0, 0, 1, 1, 1.0f);
    gemm<ACT_NONE, 1, 1, 4, 0>(stream, a_bf, mprojw_bf, m_proj_b, mh_f, nullptr,
        rowsM, ML, ML, ML, ML, ML, 0, 0, 0, 0, 0, 0, 1, 1, 1.0f);

    // mini MLP (residual fused via ACC)
    k_layernorm<<<rowsM, 256, 0, stream>>>(mh_f, m_ln2_g, m_ln2_b, mbf1, nullptr,
                                           ML, 1e-5f);
    gemm<ACT_GELU, 0, 1, 4, 1>(stream, mbf1, mw1_bf, m_mlp_b1, mh1_f, mh1_bf,
        rowsM, MF, ML, ML, ML, MF, 0, 0, 0, 0, 0, 0, 1, 1, 1.0f);
    gemm<ACT_NONE, 1, 1, 4, 0>(stream, mh1_bf, mw2_bf, m_mlp_b2, mh_f, nullptr,
        rowsM, ML, MF, MF, MF, ML, 0, 0, 0, 0, 0, 0, 1, 1, 1.0f);

    // wn^T[b,h,m,l] = softmax over l of mh[b,h,l,m]
    k_softmax_mid_t<<<dim3(1, BH), 256, 0, stream>>>(mh_f, wnT_bf, ML, ML);

    // o2[b,m,h,c] = sum_l wn[b,h,l,m] * o[b,l,h,c]  == wn^T @ (o^T)^T
    gemm<ACT_NONE, 0, 0, 2, 0>(stream, wnT_bf, opT_bf, nullptr, ao_f, nullptr,
        Lx, hd, Lx, Lx, Lx, Cc,
        (long long)Hh * Lx * Lx, (long long)Lx * Lx,
        (long long)Hh * hd * Lx, (long long)hd * Lx,
        (long long)Lx * Cc, hd, BH, Hh, 1.0f);

    // h = o2 + h ; outer MLP with fused residual
    k_add<<<dim3((unsigned)((rowsO * Cc + 255) / 256)), 256, 0, stream>>>(
        h_f, ao_f, (long long)rowsO * Cc);
    k_layernorm<<<rowsO, 256, 0, stream>>>(h_f, ln2_g + (size_t)i * Cc,
                                           ln2_b + (size_t)i * Cc, bf1, nullptr,
                                           Cc, 1e-5f);
    gemm<ACT_GELU, 0, 1, 4, 1>(stream, bf1, w1_bf, mlp_b1 + (size_t)i * Ff,
        h1_f, h1_bf,
        rowsO, Ff, Cc, Cc, Cc, Ff, 0, 0, 0, 0, 0, 0, 1, 1, 1.0f);
    gemm<ACT_NONE, 1, 1, 4, 0>(stream, h1_bf, w2_bf, mlp_b2 + (size_t)i * Cc,
        h_f, nullptr,
        rowsO, Cc, Ff, Ff, Ff, Cc, 0, 0, 0, 0, 0, 0, 1, 1, 1.0f);
  }

  // mean-pool, final LN (also f32 out), tiny ragged head GEMM on VALU
  k_pool<<<dim3((unsigned)((Bx * Cc + 255) / 256)), 256, 0, stream>>>(h_f, pool_f);
  k_layernorm<<<Bx, 256, 0, stream>>>(pool_f, norm_g, norm_b, pool_bf, pooln_f,
                                      Cc, 1e-5f);
  k_head<<<1, 128, 0, stream>>>(pooln_f, head_w, head_b, (float*)d_out);
}
